// MFFNet_30588757082247
// MI455X (gfx1250) — matrix operand order repaired and checked by reading, not compiled, not run on hardware
//
#include <hip/hip_runtime.h>

#if defined(__has_builtin)
# if __has_builtin(__builtin_amdgcn_tensor_load_to_lds) && __has_builtin(__builtin_amdgcn_s_wait_tensorcnt)
#  define HAVE_TDM 1
# endif
#endif
#ifndef HAVE_TDM
# define HAVE_TDM 0
#endif

#define DEV __device__ __forceinline__

typedef __bf16 bf16;
typedef __attribute__((ext_vector_type(16))) __bf16 v16bf;
typedef __attribute__((ext_vector_type(8)))  float  v8f;
typedef unsigned int u32x4 __attribute__((ext_vector_type(4)));
typedef int          i32x4 __attribute__((ext_vector_type(4)));
typedef int          i32x8 __attribute__((ext_vector_type(8)));

union FragBF { v16bf v; uint4 u[2]; };

DEV bf16 f2bf(float f) {
  union { float f; unsigned u; } a; a.f = f;
  unsigned r = a.u + 0x7FFFu + ((a.u >> 16) & 1u);   // round-to-nearest-even
  unsigned short h = (unsigned short)(r >> 16);
  bf16 o; __builtin_memcpy(&o, &h, sizeof(o));
  return o;
}

#define BM 128
#define BN 128
#define BKK 32
#define LDT 40   // padded LDS row stride in bf16 elems (80B = 64B data + 16B pad)

// ---------------------------------------------------------------------------
// TDM: load a BN x BKK bf16 tile (row stride K elems) into LDS at lds_off,
// padding each 64B row with 16B so the LDS row stride is LDT elems.
// D# per CDNA5 ISA ch.8: group0 {count,lds_addr,global_addr,type=2},
// group1 {data_size=2B, pad_enable, pad_interval=16DW, pad_amount=4DW,
//         tensor_dim0=K, tensor_dim1=N, tile_dim0=32, tile_dim1=128, stride0=K}.
// ---------------------------------------------------------------------------
DEV void tdm_load_b_tile(const bf16* gsrc, unsigned lds_off, int K, int N)
{
#if HAVE_TDM
  unsigned long long ga = (unsigned long long)(const void*)gsrc;
  u32x4 g0;
  g0[0] = 1u;                                               // count=1 (valid)
  g0[1] = lds_off;                                          // lds_addr
  g0[2] = (unsigned)(ga & 0xFFFFFFFFull);                   // global_addr[31:0]
  g0[3] = (unsigned)((ga >> 32) & 0x1FFFFFFull) | (2u << 30); // ga[56:32], type=2
  i32x8 g1;
  g1[0] = (1 << 16) | (1 << 20) | (3 << 22) | (3 << 25);    // 2B, pad_en, 16DW, 4DW
  g1[1] = (int)((unsigned)(K & 0xFFFF) << 16);              // tensor_dim0 lo
  g1[2] = (int)(((unsigned)K >> 16) | ((unsigned)(N & 0xFFFF) << 16));
  g1[3] = (int)(((unsigned)N >> 16) | (32u << 16));         // tile_dim0 = 32
  g1[4] = BN;                                               // tile_dim1 = 128
  g1[5] = K;                                                // tensor_dim0_stride
  g1[6] = 0; g1[7] = 0;
  i32x4 gz = {0, 0, 0, 0};
# if defined(__clang_major__) && (__clang_major__ >= 23)
  i32x8 gz8 = {0, 0, 0, 0, 0, 0, 0, 0};
  __builtin_amdgcn_tensor_load_to_lds(g0, g1, gz, gz, gz8, 0);
# else
  __builtin_amdgcn_tensor_load_to_lds(g0, g1, gz, gz, 0);
# endif
#else
  (void)gsrc; (void)lds_off; (void)K; (void)N;
#endif
}

DEV void tdm_wait0()
{
#if HAVE_TDM
  __builtin_amdgcn_s_wait_tensorcnt(0);
#endif
}

// ---------------------------------------------------------------------------
// Generic WMMA GEMM:  Y = act((X @ W^T + bias) * alpha) * coef[b]  (+ add) (+=)
//   X: [M,K] fp32; W: BTRANS ? [N,K] fp32 : [K,N] fp32
//   Wh: optional pre-converted bf16 copy of W (enables TDM pipelined path)
// Tiles: 128x128x32, 8 waves (4x2), each wave 32x64 = 2x4 WMMA 16x16 tiles.
// ---------------------------------------------------------------------------
template<int ACT, bool BTRANS, bool ACC>
__global__ __launch_bounds__(256)
void gemm_k(const float* __restrict__ X, const float* __restrict__ W,
            const bf16* __restrict__ Wh,
            const float* __restrict__ bias, const float* __restrict__ add,
            const float* __restrict__ coef, int cstride, float alpha,
            float* __restrict__ Y, int M, int N, int K,
            int in_T, int out_T, int shift,
            long long sX, long long sW, long long sY)
{
  __shared__ __align__(16) bf16 As[2][BM][LDT];
  __shared__ __align__(16) bf16 Bs[2][BN][LDT];

  const int tid  = threadIdx.x;
  const int lane = tid & 31;
  const int wave = tid >> 5;
  const int wm   = wave & 3;
  const int wn   = wave >> 2;
  const int rsel = lane & 15;
  const int hi   = (lane >> 4) & 1;

  const int n0 = blockIdx.x * BN;
  const int m0 = blockIdx.y * BM;
  const int z  = blockIdx.z;

  const int bidx = m0 / out_T;
  const float* Xb = X + (long long)z * sX
                  + ((long long)bidx * in_T + (m0 % out_T) + shift) * (long long)K;
  const float* Wb = W + (long long)z * sW;
  float* Yb = Y + (long long)z * sY;
  const float* addb = add ? add + (long long)z * sY : nullptr;
  const float cf = coef ? coef[(long long)bidx * cstride] : 1.0f;

  v8f acc[2][4];
  #pragma unroll
  for (int i = 0; i < 2; ++i)
    #pragma unroll
    for (int j = 0; j < 4; ++j)
      #pragma unroll
      for (int e = 0; e < 8; ++e) acc[i][j][e] = 0.0f;

  auto mma_step = [&](int buf) {
    FragBF af[2], bq[4];
    #pragma unroll
    for (int i = 0; i < 2; ++i) {
      const bf16* p = &As[buf][wm * 32 + i * 16 + rsel][hi * 8];
      af[i].u[0] = *reinterpret_cast<const uint4*>(p);
      af[i].u[1] = *reinterpret_cast<const uint4*>(p + 16);
    }
    #pragma unroll
    for (int j = 0; j < 4; ++j) {
      const bf16* p = &Bs[buf][wn * 64 + j * 16 + rsel][hi * 8];
      bq[j].u[0] = *reinterpret_cast<const uint4*>(p);
      bq[j].u[1] = *reinterpret_cast<const uint4*>(p + 16);
    }
    #pragma unroll
    for (int i = 0; i < 2; ++i)
      #pragma unroll
      for (int j = 0; j < 4; ++j)
        acc[i][j] = __builtin_amdgcn_wmma_f32_16x16x32_bf16(
            false, af[i].v, false, bq[j].v, (short)0, acc[i][j], false, false);
  };

  auto loadA = [&](float4 (&ra)[4], int k0) {
    #pragma unroll
    for (int r = 0; r < 4; ++r) {
      int e = (r * 256 + tid) * 4;
      int row = e >> 5, col = e & 31;
      ra[r] = *reinterpret_cast<const float4*>(Xb + (long long)row * K + k0 + col);
    }
  };
  auto storeA = [&](int buf, const float4 (&ra)[4]) {
    #pragma unroll
    for (int r = 0; r < 4; ++r) {
      int e = (r * 256 + tid) * 4;
      int row = e >> 5, col = e & 31;
      bf16* d = &As[buf][row][col];
      d[0] = f2bf(ra[r].x); d[1] = f2bf(ra[r].y);
      d[2] = f2bf(ra[r].z); d[3] = f2bf(ra[r].w);
    }
  };

  const bool fk = ((K & (BKK - 1)) == 0);
  const bool fB = fk && (n0 + BN <= N);
  const bool useTdm = HAVE_TDM && BTRANS && fB && (Wh != nullptr);

  if (useTdm) {
    // -------- pipelined path: TDM streams B tiles, double-buffered LDS ------
    const bf16* Wtile = Wh + (long long)n0 * K;     // bf16 weight rows n0..n0+127
    if (wave == 0)
      tdm_load_b_tile(Wtile, (unsigned)(unsigned long long)&Bs[0][0][0], K, N);
    float4 ra[4];
    loadA(ra, 0);
    const int steps = K / BKK;
    for (int i = 0; i < steps; ++i) {
      const int cur = i & 1;
      storeA(cur, ra);
      const bool more = (i + 1 < steps);
      if (more) loadA(ra, (i + 1) * BKK);           // overlap next A tile
      if (wave == 0) tdm_wait0();                   // B tile for step i complete
      __syncthreads();                              // As/Bs[cur] ready, readers drained
      if (more && wave == 0)                        // safe: Bs[cur^1] readers done
        tdm_load_b_tile(Wtile + (long long)(i + 1) * BKK,
                        (unsigned)(unsigned long long)&Bs[cur ^ 1][0][0], K, N);
      mma_step(cur);
    }
  } else {
    // -------- legacy path (small/irregular GEMMs) ---------------------------
    for (int k0 = 0; k0 < K; k0 += BKK) {
      if (fk) {
        float4 ra[4];
        loadA(ra, k0);
        storeA(0, ra);
        if (k0 + BKK < K)
          __builtin_prefetch(Xb + (long long)(tid >> 1) * K + k0 + BKK, 0, 1);
      } else {
        for (int r = 0; r < 16; ++r) {
          int e = r * 256 + tid;
          int row = e >> 5, col = e & 31;
          float vv = ((m0 + row) < M && (k0 + col) < K)
                   ? Xb[(long long)row * K + k0 + col] : 0.0f;
          As[0][row][col] = f2bf(vv);
        }
      }
      if (BTRANS) {
        if (fB) {
          #pragma unroll
          for (int r = 0; r < 4; ++r) {
            int e = (r * 256 + tid) * 4;
            int row = e >> 5, col = e & 31;
            float4 vv = *reinterpret_cast<const float4*>(Wb + (long long)(n0 + row) * K + k0 + col);
            bf16* d = &Bs[0][row][col];
            d[0] = f2bf(vv.x); d[1] = f2bf(vv.y); d[2] = f2bf(vv.z); d[3] = f2bf(vv.w);
          }
        } else {
          for (int r = 0; r < 16; ++r) {
            int e = r * 256 + tid;
            int row = e >> 5, col = e & 31;
            float vv = ((n0 + row) < N && (k0 + col) < K)
                     ? Wb[(long long)(n0 + row) * K + k0 + col] : 0.0f;
            Bs[0][row][col] = f2bf(vv);
          }
        }
      } else {
        if (fB) {
          #pragma unroll
          for (int r = 0; r < 4; ++r) {
            int e = (r * 256 + tid) * 4;
            int kk = e >> 7, nn = e & 127;
            float4 vv = *reinterpret_cast<const float4*>(Wb + (long long)(k0 + kk) * N + n0 + nn);
            Bs[0][nn + 0][kk] = f2bf(vv.x); Bs[0][nn + 1][kk] = f2bf(vv.y);
            Bs[0][nn + 2][kk] = f2bf(vv.z); Bs[0][nn + 3][kk] = f2bf(vv.w);
          }
        } else {
          for (int r = 0; r < 16; ++r) {
            int e = r * 256 + tid;
            int kk = e >> 7, nn = e & 127;
            float vv = ((k0 + kk) < K && (n0 + nn) < N)
                     ? Wb[(long long)(k0 + kk) * N + n0 + nn] : 0.0f;
            Bs[0][nn][kk] = f2bf(vv);
          }
        }
      }
      __syncthreads();
      mma_step(0);
      __syncthreads();
    }
  }

  // ---- epilogue ----
  const int mb = hi * 8;
  #pragma unroll
  for (int i = 0; i < 2; ++i) {
    #pragma unroll
    for (int j = 0; j < 4; ++j) {
      int n = n0 + wn * 64 + j * 16 + rsel;
      if (n >= N) continue;
      float bsv = bias ? bias[n] : 0.0f;
      #pragma unroll
      for (int v = 0; v < 8; ++v) {
        int m = m0 + wm * 32 + i * 16 + mb + v;
        if (m >= M) continue;
        float r = (acc[i][j][v] + bsv) * alpha;
        if (ACT == 1) r = fmaxf(r, 0.0f);
        else if (ACT == 2) r = 0.5f * r * (1.0f + erff(r * 0.70710678118654752f));
        else if (ACT == 3) r = 1.0f / (1.0f + __expf(-r));
        r *= cf;
        long long o = (long long)m * N + n;
        if (addb) r += addb[o];
        if (ACC) Yb[o] += r; else Yb[o] = r;
      }
    }
  }
}

// ---------------------------------------------------------------------------
__global__ __launch_bounds__(256)
void ln_k(const float* __restrict__ x, const float* __restrict__ w,
          const float* __restrict__ b, float* __restrict__ y, int D, int accum)
{
  __shared__ float red[256];
  const int row = blockIdx.x, t = threadIdx.x;
  const float* xr = x + (long long)row * D;
  float s = 0.f;
  for (int i = t; i < D; i += 256) s += xr[i];
  red[t] = s; __syncthreads();
  for (int k = 128; k > 0; k >>= 1) { if (t < k) red[t] += red[t + k]; __syncthreads(); }
  float mean = red[0] / (float)D; __syncthreads();
  float v = 0.f;
  for (int i = t; i < D; i += 256) { float d = xr[i] - mean; v += d * d; }
  red[t] = v; __syncthreads();
  for (int k = 128; k > 0; k >>= 1) { if (t < k) red[t] += red[t + k]; __syncthreads(); }
  float inv = rsqrtf(red[0] / (float)D + 1e-12f); __syncthreads();
  float* yr = y + (long long)row * D;
  for (int i = t; i < D; i += 256) {
    float r = (xr[i] - mean) * inv * w[i] + b[i];
    if (accum) yr[i] += r; else yr[i] = r;
  }
}

__global__ __launch_bounds__(256)
void softmax_k(float* __restrict__ x, int L)
{
  __shared__ float red[256];
  const int row = blockIdx.x, t = threadIdx.x;
  float* xr = x + (long long)row * L;
  float mx = -3.4e38f;
  for (int i = t; i < L; i += 256) mx = fmaxf(mx, xr[i]);
  red[t] = mx; __syncthreads();
  for (int k = 128; k > 0; k >>= 1) { if (t < k) red[t] = fmaxf(red[t], red[t + k]); __syncthreads(); }
  mx = red[0]; __syncthreads();
  float s = 0.f;
  for (int i = t; i < L; i += 256) { float e = __expf(xr[i] - mx); xr[i] = e; s += e; }
  red[t] = s; __syncthreads();
  for (int k = 128; k > 0; k >>= 1) { if (t < k) red[t] += red[t + k]; __syncthreads(); }
  float inv = 1.0f / red[0]; __syncthreads();
  for (int i = t; i < L; i += 256) xr[i] *= inv;
}

__global__ void mean_t_k(const float* f0, const float* f1, const float* f2,
                         const float* f3, const float* f4, const float* f5,
                         float* __restrict__ mu)
{
  int i = blockIdx.x * 256 + threadIdx.x;
  if (i >= 6 * 8 * 768) return;
  int n = i / (8 * 768), rem = i % (8 * 768), b = rem / 768, d = rem % 768;
  const float* f = n == 0 ? f0 : n == 1 ? f1 : n == 2 ? f2 : n == 3 ? f3 : n == 4 ? f4 : f5;
  const float* p = f + ((long long)b * 512) * 768 + d;
  float s = 0.f;
  for (int t = 0; t < 512; ++t) s += p[(long long)t * 768];
  mu[i] = s * (1.0f / 512.0f);
}

__global__ void colsum_k(const float* __restrict__ w, float* __restrict__ wbar)
{
  int i = blockIdx.x * 256 + threadIdx.x;
  if (i >= 6 * 768) return;
  int n = i / 768, e = i % 768;
  const float* p = w + (long long)n * 768 * 768 + e;
  float s = 0.f;
  for (int d = 0; d < 768; ++d) s += p[(long long)d * 768];
  wbar[i] = s;
}

__global__ void fusecoef_k(const float* __restrict__ mu, const float* __restrict__ wbar,
                           const float* __restrict__ align_b, const float* __restrict__ att_w,
                           const float* __restrict__ att_b, float* __restrict__ cvec)
{
  int b = threadIdx.x;
  if (b >= 8) return;
  float m[6];
  for (int n = 0; n < 6; ++n) {
    const float* mp = mu + ((long long)n * 8 + b) * 768;
    const float* wp = wbar + n * 768;
    float dot = 0.f;
    for (int e = 0; e < 768; ++e) dot += mp[e] * wp[e];
    float sb = 0.f;
    const float* bp = align_b + n * 768;
    for (int d = 0; d < 768; ++d) sb += bp[d];
    m[n] = (dot + sb) * (1.0f / 768.0f);
  }
  float z[6]; float mx = -3.4e38f;
  for (int o = 0; o < 6; ++o) {
    float s = att_b[o];
    for (int n = 0; n < 6; ++n) s += (att_w[o * 12 + n] + att_w[o * 12 + 6 + n]) * m[n];
    z[o] = fmaxf(s, 0.0f);
    mx = fmaxf(mx, z[o]);
  }
  float sum = 0.f;
  for (int o = 0; o < 6; ++o) { z[o] = __expf(z[o] - mx); sum += z[o]; }
  for (int o = 0; o < 6; ++o) cvec[b * 6 + o] = z[o] / sum + (1.0f / 6.0f);
}

__global__ void concat_k(const float* __restrict__ Et, const float* __restrict__ Ea,
                         float* __restrict__ cc)
{
  long long i = (long long)blockIdx.x * 256 + threadIdx.x;
  if (i >= 4096LL * 1536) return;
  long long m = i / 1536; int j = (int)(i % 1536);
  cc[i] = (j < 768) ? Et[m * 768 + j] : Ea[m * 768 + (j - 768)];
}

__global__ void gate_k(const float* __restrict__ g, const float* __restrict__ Ea,
                       const float* __restrict__ Et, float* __restrict__ Ef)
{
  long long i = (long long)blockIdx.x * 256 + threadIdx.x;
  if (i >= 4096LL * 768) return;
  float gv = g[i];
  Ef[i] = gv * Ea[i] + (1.0f - gv) * Et[i];
}

__global__ void pad_k(const float* __restrict__ x, float* __restrict__ xp)
{
  long long i = (long long)blockIdx.x * 256 + threadIdx.x;
  if (i >= 8LL * 516 * 768) return;
  int b = (int)(i / (516 * 768)); int rem = (int)(i % (516 * 768));
  int r = rem / 768, d = rem % 768;
  xp[i] = (r >= 2 && r < 514) ? x[((long long)b * 512 + (r - 2)) * 768 + d] : 0.0f;
}

__global__ void repack_k(const float* __restrict__ w, float* __restrict__ wr, int ks)
{
  long long tot = (long long)ks * 768 * 768;
  long long i = (long long)blockIdx.x * 256 + threadIdx.x;
  if (i >= tot) return;
  int j = (int)(i / (768 * 768)); int rem = (int)(i % (768 * 768));
  int o = rem / 768, e = rem % 768;
  wr[i] = w[((long long)o * 768 + e) * ks + j];
}

__global__ void cvt_bf16_k(const float* __restrict__ x, bf16* __restrict__ y, long long n)
{
  long long i = (long long)blockIdx.x * 256 + threadIdx.x;
  if (i < n) y[i] = f2bf(x[i]);
}

// ---------------------------------------------------------------------------
extern "C" void kernel_launch(void* const* d_in, const int* in_sizes, int n_in,
                              void* d_out, int out_size, void* d_ws, size_t ws_size,
                              hipStream_t stream)
{
  (void)in_sizes; (void)n_in; (void)out_size; (void)ws_size;
  const float* f[6]; for (int i = 0; i < 6; ++i) f[i] = (const float*)d_in[i];
  const float* Ea      = (const float*)d_in[6];
  const float* align_w = (const float*)d_in[7];  const float* align_b = (const float*)d_in[8];
  const float* att_w   = (const float*)d_in[9];  const float* att_b   = (const float*)d_in[10];
  const float* g1_w = (const float*)d_in[11]; const float* g1_b = (const float*)d_in[12];
  const float* g2_w = (const float*)d_in[13]; const float* g2_b = (const float*)d_in[14];
  const float* g3_w = (const float*)d_in[15]; const float* g3_b = (const float*)d_in[16];
  const float* g4_w = (const float*)d_in[17]; const float* g4_b = (const float*)d_in[18];
  const float* ln_w = (const float*)d_in[19]; const float* ln_b = (const float*)d_in[20];
  const float* c1_w = (const float*)d_in[21]; const float* c1_b = (const float*)d_in[22];
  const float* c3_w = (const float*)d_in[23]; const float* c3_b = (const float*)d_in[24];
  const float* c5_w = (const float*)d_in[25]; const float* c5_b = (const float*)d_in[26];
  const float* q_w  = (const float*)d_in[27]; const float* q_b  = (const float*)d_in[28];
  const float* qa_w = (const float*)d_in[29]; const float* qa_b = (const float*)d_in[30];
  const float* k_w  = (const float*)d_in[31]; const float* k_b  = (const float*)d_in[32];
  const float* ka_w = (const float*)d_in[33]; const float* ka_b = (const float*)d_in[34];
  const float* tr_w = (const float*)d_in[35]; const float* tr_b = (const float*)d_in[36];
  const float* ao_w = (const float*)d_in[37]; const float* ao_b = (const float*)d_in[38];
  const float* fln_w= (const float*)d_in[39]; const float* fln_b= (const float*)d_in[40];
  const float* fi_w = (const float*)d_in[41]; const float* fi_b = (const float*)d_in[42];
  const float* fo_w = (const float*)d_in[43]; const float* fo_b = (const float*)d_in[44];
  const float* m1_w = (const float*)d_in[45]; const float* m1_b = (const float*)d_in[46];
  const float* m2_w = (const float*)d_in[47]; const float* m2_b = (const float*)d_in[48];
  const float* out_w= (const float*)d_in[49]; const float* out_b= (const float*)d_in[50];

  const long long MD = 4096LL * 768;
  char* wsb = (char*)d_ws;
  size_t off = 0;
  auto allocf = [&](size_t n) { float* p = (float*)(wsb + off); off += (n * 4 + 255) & ~(size_t)255; return p; };
  auto alloch = [&](size_t n) { bf16*  p = (bf16*)(wsb + off);  off += (n * 2 + 255) & ~(size_t)255; return p; };

  float* mu   = allocf(6 * 8 * 768);
  float* wbar = allocf(6 * 768);
  float* cvec = allocf(64);
  float* w3r  = allocf(3LL * 768 * 768);
  float* w5r  = allocf(5LL * 768 * 768);
  float* Et   = allocf(MD);
  float* Ef   = allocf(MD);
  float* xb   = allocf(MD);
  float* xp   = allocf(8LL * 516 * 768);
  float* U1   = allocf(MD);
  float* U3   = allocf(MD);
  float* U5   = allocf(MD);
  float* P    = allocf(MD);
  float* bA   = allocf(MD);
  float* bB   = allocf(MD);
  float* bC   = allocf(MD);
  float* S    = allocf(8LL * 512 * 512);
  float* qsb  = allocf(4096LL * 12);
  float* ksb  = allocf(4096LL * 12);
  float* FF   = allocf(4096LL * 3072);
  // bf16 weight copies for the TDM path
  bf16* alw_h = alloch(6LL * 768 * 768);
  bf16* g1_h  = alloch(1536LL * 1536);
  bf16* g2_h  = alloch(1536LL * 1536);
  bf16* g3_h  = alloch(768LL * 1536);
  bf16* g4_h  = alloch(768LL * 768);
  bf16* c1_h  = alloch(768LL * 768);
  bf16* w3_h  = alloch(3LL * 768 * 768);
  bf16* w5_h  = alloch(5LL * 768 * 768);
  bf16* q_h   = alloch(768LL * 768);
  bf16* k_h   = alloch(768LL * 768);
  bf16* tr_h  = alloch(768LL * 768);
  bf16* ao_h  = alloch(768LL * 768);
  bf16* fi_h  = alloch(3072LL * 768);
  bf16* fo_h  = alloch(768LL * 3072);
  bf16* m1_h  = alloch(3072LL * 768);
  bf16* m2_h  = alloch(768LL * 3072);
  bf16* out_h = alloch(768LL * 768);

  auto cvt = [&](const float* src, bf16* dst, long long n) {
    cvt_bf16_k<<<(unsigned)((n + 255) / 256), 256, 0, stream>>>(src, dst, n);
  };

  auto gemm = [&](int act, bool btrans, bool accm,
                  const float* X, const float* W, const bf16* Wh,
                  const float* bias, const float* add,
                  const float* coef, int cstride, float alpha, float* Y,
                  int M, int N, int K, int in_T, int out_T, int shift,
                  long long sX, long long sW, long long sY, int batch) {
    dim3 grid((N + BN - 1) / BN, (M + BM - 1) / BM, batch), blk(256);
    if (!btrans)
      gemm_k<0,false,false><<<grid,blk,0,stream>>>(X,W,Wh,bias,add,coef,cstride,alpha,Y,M,N,K,in_T,out_T,shift,sX,sW,sY);
    else if (accm)
      gemm_k<0,true,true ><<<grid,blk,0,stream>>>(X,W,Wh,bias,add,coef,cstride,alpha,Y,M,N,K,in_T,out_T,shift,sX,sW,sY);
    else if (act == 0)
      gemm_k<0,true,false><<<grid,blk,0,stream>>>(X,W,Wh,bias,add,coef,cstride,alpha,Y,M,N,K,in_T,out_T,shift,sX,sW,sY);
    else if (act == 1)
      gemm_k<1,true,false><<<grid,blk,0,stream>>>(X,W,Wh,bias,add,coef,cstride,alpha,Y,M,N,K,in_T,out_T,shift,sX,sW,sY);
    else if (act == 2)
      gemm_k<2,true,false><<<grid,blk,0,stream>>>(X,W,Wh,bias,add,coef,cstride,alpha,Y,M,N,K,in_T,out_T,shift,sX,sW,sY);
    else
      gemm_k<3,true,false><<<grid,blk,0,stream>>>(X,W,Wh,bias,add,coef,cstride,alpha,Y,M,N,K,in_T,out_T,shift,sX,sW,sY);
  };

  // ---- one-time fp32 -> bf16 weight conversion (feeds TDM B-tile loads) ----
  cvt(align_w, alw_h, 6LL * 768 * 768);
  cvt(g1_w, g1_h, 1536LL * 1536);
  cvt(g2_w, g2_h, 1536LL * 1536);
  cvt(g3_w, g3_h, 768LL * 1536);
  cvt(g4_w, g4_h, 768LL * 768);
  cvt(c1_w, c1_h, 768LL * 768);
  cvt(q_w,  q_h,  768LL * 768);
  cvt(k_w,  k_h,  768LL * 768);
  cvt(tr_w, tr_h, 768LL * 768);
  cvt(ao_w, ao_h, 768LL * 768);
  cvt(fi_w, fi_h, 3072LL * 768);
  cvt(fo_w, fo_h, 768LL * 3072);
  cvt(m1_w, m1_h, 3072LL * 768);
  cvt(m2_w, m2_h, 768LL * 3072);
  cvt(out_w, out_h, 768LL * 768);

  // ---- fusion coefficients (tiny; uses full-mean == mean_T.mean_D identity) ----
  mean_t_k<<<(6 * 8 * 768 + 255) / 256, 256, 0, stream>>>(f[0], f[1], f[2], f[3], f[4], f[5], mu);
  colsum_k<<<(6 * 768 + 255) / 256, 256, 0, stream>>>(align_w, wbar);
  fusecoef_k<<<1, 64, 0, stream>>>(mu, wbar, align_b, att_w, att_b, cvec);

  // ---- Et = sum_n c[b,n] * (f_n @ Wn^T + bn) ----
  for (int n = 0; n < 6; ++n)
    gemm(0, true, n > 0, f[n], align_w + (size_t)n * 768 * 768, alw_h + (size_t)n * 768 * 768,
         align_b + n * 768, nullptr, cvec + n, 6, 1.0f, Et, 4096, 768, 768, 512, 512, 0, 0, 0, 0, 1);

  // ---- GatedFusion ----
  float* cc = FF; float* o1 = FF + 4096LL * 1536; float* o2 = FF;
  concat_k<<<(unsigned)((4096LL * 1536 + 255) / 256), 256, 0, stream>>>(Et, Ea, cc);
  gemm(1, true, false, cc, g1_w, g1_h, g1_b, nullptr, nullptr, 0, 1.0f, o1, 4096, 1536, 1536, 512, 512, 0, 0, 0, 0, 1);
  gemm(1, true, false, o1, g2_w, g2_h, g2_b, nullptr, nullptr, 0, 1.0f, o2, 4096, 1536, 1536, 512, 512, 0, 0, 0, 0, 1);
  gemm(1, true, false, o2, g3_w, g3_h, g3_b, nullptr, nullptr, 0, 1.0f, bA, 4096, 768, 1536, 512, 512, 0, 0, 0, 0, 1);
  gemm(3, true, false, bA, g4_w, g4_h, g4_b, nullptr, nullptr, 0, 1.0f, bB, 4096, 768, 768, 512, 512, 0, 0, 0, 0, 1);
  gate_k<<<(unsigned)((4096LL * 768 + 255) / 256), 256, 0, stream>>>(bB, Ea, Et, Ef);

  // ---- MSFastformer front: LN + multi-scale convs as shifted GEMMs ----
  ln_k<<<4096, 256, 0, stream>>>(Ef, ln_w, ln_b, xb, 768, 0);
  pad_k<<<(unsigned)((8LL * 516 * 768 + 255) / 256), 256, 0, stream>>>(xb, xp);
  repack_k<<<(unsigned)((3LL * 768 * 768 + 255) / 256), 256, 0, stream>>>(c3_w, w3r, 3);
  repack_k<<<(unsigned)((5LL * 768 * 768 + 255) / 256), 256, 0, stream>>>(c5_w, w5r, 5);
  cvt(w3r, w3_h, 3LL * 768 * 768);
  cvt(w5r, w5_h, 5LL * 768 * 768);

  gemm(0, true, false, xb, c1_w, c1_h, c1_b, nullptr, nullptr, 0, 1.0f, U1, 4096, 768, 768, 512, 512, 0, 0, 0, 0, 1);
  for (int j = 0; j < 3; ++j)
    gemm(0, true, j > 0, xp, w3r + (size_t)j * 768 * 768, w3_h + (size_t)j * 768 * 768,
         j == 0 ? c3_b : nullptr, nullptr, nullptr, 0, 1.0f, U3, 4096, 768, 768, 516, 512, j + 1, 0, 0, 0, 1);
  for (int j = 0; j < 5; ++j)
    gemm(0, true, j > 0, xp, w5r + (size_t)j * 768 * 768, w5_h + (size_t)j * 768 * 768,
         j == 0 ? c5_b : nullptr, nullptr, nullptr, 0, 1.0f, U5, 4096, 768, 768, 516, 512, j, 0, 0, 0, 1);

  // ---- Fastformer layer (shared weights), accumulated into P ----
  const float scale = 0.125f;  // 1/sqrt(64)
  auto fastformer = [&](const float* hid, const float* ctx, int accP) {
    gemm(0, true, false, hid, q_w, q_h, q_b, nullptr, nullptr, 0, 1.0f, bA, 4096, 768, 768, 512, 512, 0, 0, 0, 0, 1);
    gemm(0, true, false, ctx, k_w, k_h, k_b, nullptr, nullptr, 0, 1.0f, bB, 4096, 768, 768, 512, 512, 0, 0, 0, 0, 1);
    gemm(0, true, false, bA, qa_w, nullptr, qa_b, nullptr, nullptr, 0, scale, qsb, 4096, 12, 768, 512, 512, 0, 0, 0, 0, 1);
    gemm(0, true, false, bB, ka_w, nullptr, ka_b, nullptr, nullptr, 0, scale, ksb, 4096, 12, 768, 512, 512, 0, 0, 0, 0, 1);
    gemm(0, true, false, qsb, ksb, nullptr, nullptr, nullptr, nullptr, 0, 1.0f, S,
         512, 512, 12, 512, 512, 0, 512LL * 12, 512LL * 12, 512LL * 512, 8);
    softmax_k<<<4096, 256, 0, stream>>>(S, 512);
    gemm(0, false, false, S, bB, nullptr, nullptr, nullptr, nullptr, 0, 1.0f, bC,
         512, 768, 512, 512, 512, 0, 512LL * 512, 512LL * 768, 512LL * 768, 8);
    gemm(0, true, false, bC, tr_w, tr_h, tr_b, bA, nullptr, 0, 1.0f, bA, 4096, 768, 768, 512, 512, 0, 0, 0, 0, 1);
    gemm(0, true, false, bA, ao_w, ao_h, ao_b, hid, nullptr, 0, 1.0f, bC, 4096, 768, 768, 512, 512, 0, 0, 0, 0, 1);
    ln_k<<<4096, 256, 0, stream>>>(bC, fln_w, fln_b, bB, 768, 0);
    gemm(2, true, false, bB, fi_w, fi_h, fi_b, nullptr, nullptr, 0, 1.0f, FF, 4096, 3072, 768, 512, 512, 0, 0, 0, 0, 1);
    gemm(0, true, false, FF, fo_w, fo_h, fo_b, bB, nullptr, 0, 1.0f, bC, 4096, 768, 3072, 512, 512, 0, 0, 0, 0, 1);
    ln_k<<<4096, 256, 0, stream>>>(bC, fln_w, fln_b, P, 768, accP);
  };
  fastformer(U5, U3, 0);
  fastformer(U3, U1, 1);
  fastformer(U1, U5, 1);

  // ---- tail: Pn -> FFN -> +x -> LN -> out projection ----
  ln_k<<<4096, 256, 0, stream>>>(P, ln_w, ln_b, bC, 768, 0);
  gemm(2, true, false, bC, m1_w, m1_h, m1_b, nullptr, nullptr, 0, 1.0f, FF, 4096, 3072, 768, 512, 512, 0, 0, 0, 0, 1);
  gemm(0, true, false, FF, m2_w, m2_h, m2_b, xb, nullptr, 0, 1.0f, bA, 4096, 768, 3072, 512, 512, 0, 0, 0, 0, 1);
  ln_k<<<4096, 256, 0, stream>>>(bA, ln_w, ln_b, bC, 768, 0);
  gemm(0, true, false, bC, out_w, out_h, out_b, nullptr, nullptr, 0, 1.0f, (float*)d_out,
       4096, 768, 768, 512, 512, 0, 0, 0, 0, 1);
}
